// Recurrence_42949673249
// MI455X (gfx1250) — matrix operand order repaired and checked by reading, not compiled, not run on hardware
//
#include <hip/hip_runtime.h>
#include <hip/hip_bf16.h>
#include <stdint.h>

// ---------------------------------------------------------------------------
// Problem constants (from reference): NL=32, NE=4, TES=128, HID=256, NA=32,
// OBSD=128, VOCAB=1000, T=64, N=256
// ---------------------------------------------------------------------------
#define C_NL   32
#define C_NE   4
#define C_TES  128
#define C_HID  256
#define C_NA   32
#define C_OBSD 128
#define C_T    64
#define C_N    256

typedef __attribute__((ext_vector_type(16))) _Float16 v16h;
typedef __attribute__((ext_vector_type(8)))  float    v8f;

// Branch-free activations: v_exp_f32 + v_rcp_f32, no precise-div sequence.
__device__ __forceinline__ float sigmoidf_(float x) {
  return __builtin_amdgcn_rcpf(1.f + __expf(-x));
}
__device__ __forceinline__ float tanhf_(float x) {
  // tanh(x) = 2*sigmoid(2x) - 1
  return 2.f * __builtin_amdgcn_rcpf(1.f + __expf(-2.f * x)) - 1.f;
}

// Optimization barrier: an opaque zero offset (SGPR) added to a weight
// pointer INSIDE the recurrent loop. The asm executes every iteration, so the
// computed addresses are "new" each step -> loop-invariant weight-tile loads
// cannot be hoisted out of the time loop (which previously caused massive
// VGPR spills to scratch). Unlike laundering the pointer itself through
// inline asm, adding an offset preserves the addrspace(1) provenance of the
// kernel argument, so the loads stay global_load_b128 (not flat_load).
__device__ __forceinline__ size_t opaque_off() {
  unsigned long long x = 0;
  asm volatile("" : "+s"(x));
  return (size_t)x;
}

// ---------------------------------------------------------------------------
// WMMA operand loaders (CDNA5 v_wmma_f32_16x16x32_f16 layouts, wave32)
// A (16x32 f16): lane L -> row M = L&15 ; halves 0..7 = K (L<16?0:8)+i,
//                halves 8..15 = K (L<16?16:24)+i   (row-major source, pitch in halves)
// B (32x16 f16): lane L -> N = L&15 ; halves 0..7 = K (L<16?0:8)+i, halves 8..15 = K (L<16?16:24)+i ([K][O] packed)
// C/D (16x16 f32): reg r -> M = r + (L<16?0:8), N = L&15
// ---------------------------------------------------------------------------
__device__ __forceinline__ v16h load_matA(const _Float16* p, int pitch) {
  const int lane = threadIdx.x & 31;
  const _Float16* r = p + (lane & 15) * pitch + ((lane & 16) ? 8 : 0);
  v16h a;
#pragma unroll
  for (int i = 0; i < 8; ++i) a[i] = r[i];
#pragma unroll
  for (int i = 0; i < 8; ++i) a[8 + i] = r[16 + i];
  return a;
}

__device__ __forceinline__ v16h load_matB(const _Float16* Wt, int O, int kbase, int obase) {
  const int lane = threadIdx.x & 31;
  const int kh = (lane & 16) ? 8 : 0;
  const _Float16* r = Wt + (size_t)kbase * O + obase + (lane & 15);
  v16h b;
#pragma unroll
  for (int i = 0; i < 8; ++i) b[i] = r[(size_t)(kh + i) * O];
#pragma unroll
  for (int i = 0; i < 8; ++i) b[8 + i] = r[(size_t)(16 + kh + i) * O];
  return b;
}

__device__ __forceinline__ v8f wmma_f16(v16h a, v16h b, v8f c) {
  // 8 args: (neg_a, A, neg_b, B, c_mod, C, reuse_a, reuse_b)
  return __builtin_amdgcn_wmma_f32_16x16x32_f16(false, a, false, b, (short)0, c, false, false);
}

// ---------------------------------------------------------------------------
// Pack f32 weight -> f16 in [rows_out(K)][cols_out(O)] layout, zero padded.
// transpose=1: dst[r][c] = src[c][r]  (for x @ W.T, src is [O][K])
// transpose=0: dst[r][c] = src[r][c]  (for x @ W,   src is [K][O])
// ---------------------------------------------------------------------------
__global__ void pack_kernel(const float* __restrict__ src, _Float16* __restrict__ dst,
                            int rows_out, int cols_out, int src_rows, int src_cols,
                            int transpose) {
  int idx = blockIdx.x * blockDim.x + threadIdx.x;
  if (idx >= rows_out * cols_out) return;
  int r = idx / cols_out, c = idx % cols_out;
  float v = 0.f;
  if (transpose) { if (c < src_rows && r < src_cols) v = src[(size_t)c * src_cols + r]; }
  else           { if (r < src_rows && c < src_cols) v = src[(size_t)r * src_cols + c]; }
  dst[idx] = (_Float16)v;
}

// ---------------------------------------------------------------------------
// Embed + roll:  X[(i*N+n)*32 + t][f] = embed_W[lines[n][(t+i)&31]][f]  (f16)
// grid = 8192*32 blocks, 128 threads
// ---------------------------------------------------------------------------
__global__ void embed_roll_kernel(const int* __restrict__ lines,
                                  const float* __restrict__ embedW,
                                  _Float16* __restrict__ X) {
  const int q = blockIdx.x;           // 0 .. 262143
  const int f = threadIdx.x;          // 0 .. 127
  const int t = q & 31;
  const int rr = q >> 5;              // i*256 + n
  const int n = rr & 255;
  const int i = rr >> 8;
  const int l = (t + i) & 31;
  const int line = lines[n * C_NL + l];
  X[(size_t)q * C_TES + f] = (_Float16)embedW[(size_t)line * C_TES + f];
}

// ---------------------------------------------------------------------------
// Bidirectional GRU scan (one direction per launch).
// 128 WGs x 64 batch rows, 1024 threads = 32 waves.
// Wave w: row-tile rt=w>>3 (16 rows), hidden col-tile ct=w&7 (16 of 128 cols).
// Per step: gi = x@WihT, gh = h@WhhT fused into 6 accumulators, combine in-reg.
// B tiles are re-fetched from L2 every step via opaque_off() to avoid spills.
// ---------------------------------------------------------------------------
__global__ __launch_bounds__(1024) void gru_scan_kernel(
    const _Float16* __restrict__ X,      // [8192][32][128] f16
    const _Float16* __restrict__ WihT,   // [128][384] f16
    const _Float16* __restrict__ WhhT,   // [128][384] f16
    const float* __restrict__ bih, const float* __restrict__ bhh,
    _Float16* __restrict__ G,            // [8192][32][256] f16
    int t0, int dt, int dirofs) {
  __shared__ __align__(16) _Float16 xT[64][128];
  __shared__ __align__(16) _Float16 hT[64][128];
  const int tid = threadIdx.x, lane = tid & 31, w = tid >> 5;
  const int rt = w >> 3, ct = w & 7;
  const int rowbase = blockIdx.x * 64;
  const int jg = ct * 16 + (lane & 15);             // hidden col 0..127
  const int lr = rt * 16 + ((lane & 16) ? 8 : 0);   // row base for C-layout
  for (int i = tid; i < 64 * 128; i += 1024) ((_Float16*)hT)[i] = (_Float16)0.f;
  float hprev[8];
#pragma unroll
  for (int r = 0; r < 8; ++r) hprev[r] = 0.f;
  const float br_i = bih[jg], bz_i = bih[128 + jg], bn_i = bih[256 + jg];
  const float br_h = bhh[jg], bz_h = bhh[128 + jg], bn_h = bhh[256 + jg];
  __syncthreads();
  for (int s = 0; s < 32; ++s) {
    const int tt = t0 + dt * s;
    {   // stage x tile: 64 rows x 128 halves = 1024 x 16B
      const int row = tid >> 4, ch = tid & 15;
      const uint4* src = reinterpret_cast<const uint4*>(X) +
                         ((size_t)(rowbase + row) * 32 + tt) * 16 + ch;
      reinterpret_cast<uint4*>(&xT[0][0])[row * 16 + ch] = *src;
    }
    __syncthreads();
    const size_t oo = opaque_off();      // defeat LICM, keep addrspace(1)
    const _Float16* Wi = WihT + oo;
    const _Float16* Wh = WhhT + oo;
    v8f gir = {}, giz = {}, gin = {}, ghr = {}, ghz = {}, ghn = {};
#pragma unroll 1
    for (int kc = 0; kc < 4; ++kc) {
      v16h ax = load_matA(&xT[rt * 16][kc * 32], 128);
      v16h ah = load_matA(&hT[rt * 16][kc * 32], 128);
      v16h b;
      b = load_matB(Wi, 384, kc * 32, 0   + ct * 16); gir = wmma_f16(ax, b, gir);
      b = load_matB(Wi, 384, kc * 32, 128 + ct * 16); giz = wmma_f16(ax, b, giz);
      b = load_matB(Wi, 384, kc * 32, 256 + ct * 16); gin = wmma_f16(ax, b, gin);
      b = load_matB(Wh, 384, kc * 32, 0   + ct * 16); ghr = wmma_f16(ah, b, ghr);
      b = load_matB(Wh, 384, kc * 32, 128 + ct * 16); ghz = wmma_f16(ah, b, ghz);
      b = load_matB(Wh, 384, kc * 32, 256 + ct * 16); ghn = wmma_f16(ah, b, ghn);
    }
    __syncthreads();   // all reads of hT complete before overwrite
#pragma unroll
    for (int r = 0; r < 8; ++r) {
      float rr = sigmoidf_((gir[r] + br_i) + (ghr[r] + br_h));
      float zz = sigmoidf_((giz[r] + bz_i) + (ghz[r] + bz_h));
      float nn = tanhf_((gin[r] + bn_i) + rr * (ghn[r] + bn_h));
      float hn2 = (1.f - zz) * nn + zz * hprev[r];
      hprev[r] = hn2;
      hT[lr + r][jg] = (_Float16)hn2;
      G[((size_t)(rowbase + lr + r) * 32 + tt) * 256 + dirofs + jg] = (_Float16)hn2;
    }
    __syncthreads();
  }
}

// ---------------------------------------------------------------------------
// Beta MLP: Bout = sigmoid(relu(G@W1+b1)@W2+b2), rows = 524288, fused.
// 4096 WGs x 128 rows, 512 threads = 16 waves.
// ---------------------------------------------------------------------------
__global__ __launch_bounds__(512) void beta_kernel(
    const _Float16* __restrict__ G16,   // [524288][128]
    const _Float16* __restrict__ W1t,   // [128][128]
    const float* __restrict__ b1,
    const _Float16* __restrict__ W2t,   // [128][16] (cols 4..15 zero)
    const float* __restrict__ b2,
    float* __restrict__ Bout) {         // [524288][4]
  __shared__ __align__(16) _Float16 h1[128][128];
  const int tid = threadIdx.x, lane = tid & 31, w = tid >> 5;
  const size_t qbase = (size_t)blockIdx.x * 128;
  const int rt = w >> 1, cg = w & 1;
  const int mrow = (lane & 16) ? 8 : 0;
  v8f acc[4] = {{}, {}, {}, {}};
#pragma unroll
  for (int kc = 0; kc < 4; ++kc) {
    v16h a = load_matA(G16 + (qbase + rt * 16) * 128 + kc * 32, 128);
#pragma unroll
    for (int ci = 0; ci < 4; ++ci) {
      v16h b = load_matB(W1t, 128, kc * 32, (cg * 4 + ci) * 16);
      acc[ci] = wmma_f16(a, b, acc[ci]);
    }
  }
#pragma unroll
  for (int ci = 0; ci < 4; ++ci) {
    int col = (cg * 4 + ci) * 16 + (lane & 15);
    float bb = b1[col];
#pragma unroll
    for (int r = 0; r < 8; ++r) {
      float v = acc[ci][r] + bb;
      h1[rt * 16 + mrow + r][col] = (_Float16)(v > 0.f ? v : 0.f);
    }
  }
  __syncthreads();
  if (w < 8) {
    v8f a2 = {};
#pragma unroll
    for (int kc = 0; kc < 4; ++kc) {
      v16h a = load_matA(&h1[w * 16][kc * 32], 128);
      v16h b = load_matB(W2t, 16, kc * 32, 0);
      a2 = wmma_f16(a, b, a2);
    }
    int n = lane & 15;
    if (n < 4) {
      float bb = b2[n];
#pragma unroll
      for (int r = 0; r < 8; ++r) {
        size_t q = qbase + w * 16 + mrow + r;
        Bout[q * 4 + n] = sigmoidf_(a2[r] + bb);
      }
    }
  }
}

// ---------------------------------------------------------------------------
// Mask / cumprod / flip -> P[32][256][64][4].  One thread per (l,n,e).
// ---------------------------------------------------------------------------
__global__ void pmask_kernel(const float* __restrict__ Bbuf, float* __restrict__ P) {
  int idx = blockIdx.x * blockDim.x + threadIdx.x;   // 32768
  int e = idx & 3;
  int ln = idx >> 2;                                 // l*256 + n
  const float* Bp = Bbuf + (size_t)ln * 32 * 2 * 4;
  float* Pp = P + (size_t)ln * 64 * 4;
  float prev = 0.f, c = 1.f;
  for (int k = 0; k < 64; ++k) {
    int t = k >> 1, s = k & 1;
    float bv = s ? Bp[((31 - t) * 2 + 1) * 4 + e] : Bp[(t * 2) * 4 + e];
    float bm = (k == 0)  ? 0.f : bv;    // firstm zeroes k=0
    float zl = (k == 63) ? 0.f : bm;    // zero_last
    float bf = (k == 63) ? 1.f : bm;    // + lastm
    c *= (1.f - prev);                  // cumprod of (1 - roll(zero_last,1))
    float pv = bf * c;
    prev = zl;
    int kk = s ? (31 - t) : (32 + t);   // concat([flip(b2), f2])
    Pp[kk * 4 + e] = pv;
  }
}

// ---------------------------------------------------------------------------
// Decision loop: 16 persistent WGs x 16 envs, T=64 steps, 512 thr = 16 waves.
// Wave w owns hidden col-tile w (GRU cell + zeta); waves 0..3 run heads;
// wave w finishes env w (softmaxes, d_probs, value, p update).
// ---------------------------------------------------------------------------
__global__ __launch_bounds__(512) void step_loop_kernel(
    const int* __restrict__ lines, const float* __restrict__ obs,
    const int* __restrict__ D, const float* __restrict__ embedW,
    const _Float16* __restrict__ cWihT,   // [160][768]
    const _Float16* __restrict__ cWhhT,   // [256][768]
    const float* __restrict__ cbih, const float* __restrict__ cbhh,
    const _Float16* __restrict__ zetaT,   // [384][256]
    const float* __restrict__ zeta_b,
    const _Float16* __restrict__ actT,    // [256][32]
    const float* __restrict__ act_b,
    const _Float16* __restrict__ upsT,    // [256][16]
    const float* __restrict__ ups_b,
    const _Float16* __restrict__ critT,   // [256][16]
    const float* __restrict__ crit_b,
    const float* __restrict__ P,          // [32][256][64][4]
    float* __restrict__ out) {            // [64][256][97]
  __shared__ __align__(16) _Float16 x1[16][160];
  __shared__ __align__(16) _Float16 hF[16][256];
  __shared__ __align__(16) _Float16 x2[16][384];
  __shared__ __align__(16) _Float16 zF[16][256];
  __shared__ float act_l[16][32];
  __shared__ float ups_l[16][16];
  __shared__ float crit_l[16][16];
  __shared__ float u_s[16][4];
  __shared__ int   p_s[16];
  const int tid = threadIdx.x, lane = tid & 31, w = tid >> 5;
  const int egBase = blockIdx.x * 16;
  for (int i = tid; i < 16 * 256; i += 512) ((_Float16*)hF)[i] = (_Float16)0.f;
  for (int i = tid; i < 16 * 160; i += 512) ((_Float16*)x1)[i] = (_Float16)0.f;
  if (tid < 64) u_s[tid >> 2][tid & 3] = 0.f;
  if (tid < 16) p_s[tid] = 0;
  const int jg = w * 16 + (lane & 15);   // hidden col 0..255
  const int mrow = (lane & 16) ? 8 : 0;
  const float br_i = cbih[jg], bz_i = cbih[256 + jg], bn_i = cbih[512 + jg];
  const float br_h = cbhh[jg], bz_h = cbhh[256 + jg], bn_h = cbhh[512 + jg];
  float hprev[8];
#pragma unroll
  for (int r = 0; r < 8; ++r) hprev[r] = 0.f;
  __syncthreads();
  for (int t = 0; t < C_T; ++t) {
    // x1[:,0:128] = embed gather via p ; x2[:,0:128] = obs_t
    for (int i = tid; i < 16 * 128; i += 512) {
      int e = i >> 7, f = i & 127;
      int line = lines[(egBase + e) * C_NL + p_s[e]];
      x1[e][f] = (_Float16)embedW[(size_t)line * C_TES + f];
      x2[e][f] = (_Float16)obs[((size_t)t * C_N + egBase + e) * C_OBSD + f];
    }
    for (int i = tid; i < 64; i += 512) { int e = i >> 2, j = i & 3;
      x1[e][128 + j] = (_Float16)u_s[e][j]; }
    __syncthreads();
    // GRU cell: gi (K=160) + gh (K=256) -> 768 gates, wave w -> col tile w
    const size_t oo = opaque_off();
    const _Float16* Wi = cWihT + oo;
    const _Float16* Wh = cWhhT + oo;
    v8f gir = {}, giz = {}, gin = {}, ghr = {}, ghz = {}, ghn = {};
#pragma unroll 1
    for (int kc = 0; kc < 5; ++kc) {
      v16h a = load_matA(&x1[0][kc * 32], 160);
      v16h b;
      b = load_matB(Wi, 768, kc * 32, 0   + w * 16); gir = wmma_f16(a, b, gir);
      b = load_matB(Wi, 768, kc * 32, 256 + w * 16); giz = wmma_f16(a, b, giz);
      b = load_matB(Wi, 768, kc * 32, 512 + w * 16); gin = wmma_f16(a, b, gin);
    }
#pragma unroll 1
    for (int kc = 0; kc < 8; ++kc) {
      v16h a = load_matA(&hF[0][kc * 32], 256);
      v16h b;
      b = load_matB(Wh, 768, kc * 32, 0   + w * 16); ghr = wmma_f16(a, b, ghr);
      b = load_matB(Wh, 768, kc * 32, 256 + w * 16); ghz = wmma_f16(a, b, ghz);
      b = load_matB(Wh, 768, kc * 32, 512 + w * 16); ghn = wmma_f16(a, b, ghn);
    }
    __syncthreads();
#pragma unroll
    for (int r = 0; r < 8; ++r) {
      float rr = sigmoidf_((gir[r] + br_i) + (ghr[r] + br_h));
      float zz = sigmoidf_((giz[r] + bz_i) + (ghz[r] + bz_h));
      float nn = tanhf_((gin[r] + bn_i) + rr * (ghn[r] + bn_h));
      float hn2 = (1.f - zz) * nn + zz * hprev[r];
      hprev[r] = hn2;
      hF[mrow + r][jg] = (_Float16)hn2;
      x2[mrow + r][128 + jg] = (_Float16)hn2;
    }
    __syncthreads();
    // z = relu([obs,h] @ zeta2_W + b): K=384, wave w -> col tile w
    const _Float16* Zt = zetaT + opaque_off();
    v8f az = {};
#pragma unroll 1
    for (int kc = 0; kc < 12; ++kc) {
      v16h a = load_matA(&x2[0][kc * 32], 384);
      v16h b = load_matB(Zt, 256, kc * 32, w * 16);
      az = wmma_f16(a, b, az);
    }
    {
      float bb = zeta_b[jg];
#pragma unroll
      for (int r = 0; r < 8; ++r) {
        float v = az[r] + bb;
        zF[mrow + r][jg] = (_Float16)(v > 0.f ? v : 0.f);
      }
    }
    __syncthreads();
    // heads: waves 0,1 -> actor[0:16),[16:32); wave 2 -> ups; wave 3 -> crit
    // opaque offset applied to the UNIFORM base pointers; the divergent select
    // happens afterwards (addrspace(1) preserved on both sides).
    {
      const size_t ho = opaque_off();
      const _Float16* aT = actT + ho;
      const _Float16* uT = upsT + ho;
      const _Float16* cT = critT + ho;
      if (w < 4) {
        const _Float16* Bp = (w < 2) ? aT : ((w == 2) ? uT : cT);
        const int O = (w < 2) ? 32 : 16;
        const int ob = (w == 1) ? 16 : 0;
        v8f ah = {};
#pragma unroll 1
        for (int kc = 0; kc < 8; ++kc) {
          v16h a = load_matA(&zF[0][kc * 32], 256);
          v16h b = load_matB(Bp, O, kc * 32, ob);
          ah = wmma_f16(a, b, ah);
        }
        int n = lane & 15;
#pragma unroll
        for (int r = 0; r < 8; ++r) {
          int e = mrow + r;
          if (w < 2)       act_l[e][ob + n] = ah[r] + act_b[ob + n];
          else if (w == 2) ups_l[e][n] = ah[r] + (n < 4 ? ups_b[n] : 0.f);
          else             crit_l[e][n] = ah[r];
        }
      }
    }
    __syncthreads();
    // per-env finish: wave w -> env w
    {
      const int e = w, eg = egBase + e;
      float* orow = out + ((size_t)t * C_N + eg) * 97;
      // actor softmax over 32 lanes
      float av = act_l[e][lane];
      float mx = av;
#pragma unroll
      for (int off = 16; off > 0; off >>= 1) {
        float o = __shfl_xor(mx, off, 32); mx = mx > o ? mx : o; }
      float ex = __expf(av - mx);
      float sm = ex;
#pragma unroll
      for (int off = 16; off > 0; off >>= 1) sm += __shfl_xor(sm, off, 32);
      float inv_sm = __builtin_amdgcn_rcpf(sm);
      orow[lane] = ex * inv_sm;
      // ups softmax (4) computed redundantly per lane
      float u0 = ups_l[e][0], u1 = ups_l[e][1], u2 = ups_l[e][2], u3 = ups_l[e][3];
      float um = fmaxf(fmaxf(u0, u1), fmaxf(u2, u3));
      float e0 = __expf(u0 - um), e1 = __expf(u1 - um),
            e2 = __expf(u2 - um), e3 = __expf(u3 - um);
      float inv_us = __builtin_amdgcn_rcpf(e0 + e1 + e2 + e3);
      float uu0 = e0 * inv_us, uu1 = e1 * inv_us, uu2 = e2 * inv_us, uu3 = e3 * inv_us;
      if (lane == 0) { u_s[e][0] = uu0; u_s[e][1] = uu1; u_s[e][2] = uu2; u_s[e][3] = uu3; }
      // d_probs = einsum(P[p, eg], u_new)
      const int pcur = p_s[e];
      const float* Pr = P + ((size_t)pcur * C_N + eg) * 64 * 4;
#pragma unroll
      for (int hh = 0; hh < 2; ++hh) {
        int kk = lane + 32 * hh;
        const float* pk = Pr + kk * 4;
        orow[32 + kk] = pk[0] * uu0 + pk[1] * uu1 + pk[2] * uu2 + pk[3] * uu3;
      }
      if (lane == 0) {
        orow[96] = crit_l[e][0] + crit_b[0];
        int pn = pcur + D[t * C_N + eg] - C_NL;
        p_s[e] = pn < 0 ? 0 : (pn > C_NL - 1 ? C_NL - 1 : pn);
      }
    }
    __syncthreads();
  }
}

// ---------------------------------------------------------------------------
extern "C" void kernel_launch(void* const* d_in, const int* in_sizes, int n_in,
                              void* d_out, int out_size, void* d_ws, size_t ws_size,
                              hipStream_t stream) {
  (void)in_sizes; (void)n_in; (void)out_size; (void)ws_size;
  const int*   lines   = (const int*)  d_in[0];
  const float* obs     = (const float*)d_in[1];
  const int*   D       = (const int*)  d_in[2];
  const float* embedW  = (const float*)d_in[3];
  const float* gWih_f  = (const float*)d_in[4];
  const float* gWhh_f  = (const float*)d_in[5];
  const float* gbih_f  = (const float*)d_in[6];
  const float* gbhh_f  = (const float*)d_in[7];
  const float* gWih_b  = (const float*)d_in[8];
  const float* gWhh_b  = (const float*)d_in[9];
  const float* gbih_b  = (const float*)d_in[10];
  const float* gbhh_b  = (const float*)d_in[11];
  const float* beta_W1 = (const float*)d_in[12];
  const float* beta_b1 = (const float*)d_in[13];
  const float* beta_W2 = (const float*)d_in[14];
  const float* beta_b2 = (const float*)d_in[15];
  const float* cWih    = (const float*)d_in[16];
  const float* cWhh    = (const float*)d_in[17];
  const float* cbih    = (const float*)d_in[18];
  const float* cbhh    = (const float*)d_in[19];
  const float* zeta_W  = (const float*)d_in[20];
  const float* zeta_b  = (const float*)d_in[21];
  const float* act_W   = (const float*)d_in[22];
  const float* act_b   = (const float*)d_in[23];
  const float* ups_W   = (const float*)d_in[24];
  const float* ups_b   = (const float*)d_in[25];
  const float* crit_W  = (const float*)d_in[26];
  const float* crit_b  = (const float*)d_in[27];
  float* out = (float*)d_out;

  char* ws = (char*)d_ws;
  size_t off = 0;
  auto carve = [&](size_t bytes) -> char* {
    char* p = ws + off; off = (off + bytes + 255) & ~(size_t)255; return p;
  };
  _Float16* X     = (_Float16*)carve((size_t)8192 * 32 * 128 * 2);   // rolled f16
  _Float16* G16   = (_Float16*)carve((size_t)8192 * 32 * 256 * 2);   // GRU outputs f16
  float*    Bbuf  = (float*)   carve((size_t)524288 * 4 * 4);        // beta sigmoid
  float*    Pbuf  = (float*)   carve((size_t)8192 * 64 * 4 * 4);     // P
  _Float16* WihFt = (_Float16*)carve(128 * 384 * 2);
  _Float16* WhhFt = (_Float16*)carve(128 * 384 * 2);
  _Float16* WihBt = (_Float16*)carve(128 * 384 * 2);
  _Float16* WhhBt = (_Float16*)carve(128 * 384 * 2);
  _Float16* W1t   = (_Float16*)carve(128 * 128 * 2);
  _Float16* W2t   = (_Float16*)carve(128 * 16 * 2);
  _Float16* cWihT = (_Float16*)carve(160 * 768 * 2);
  _Float16* cWhhT = (_Float16*)carve(256 * 768 * 2);
  _Float16* zetaT = (_Float16*)carve(384 * 256 * 2);
  _Float16* actT  = (_Float16*)carve(256 * 32 * 2);
  _Float16* upsT  = (_Float16*)carve(256 * 16 * 2);
  _Float16* critT = (_Float16*)carve(256 * 16 * 2);

  auto pack = [&](const float* src, _Float16* dst, int ro, int co, int sr, int sc, int tr) {
    int total = ro * co;
    pack_kernel<<<(total + 255) / 256, 256, 0, stream>>>(src, dst, ro, co, sr, sc, tr);
  };
  pack(gWih_f,  WihFt, 128, 384, 384, 128, 1);
  pack(gWhh_f,  WhhFt, 128, 384, 384, 128, 1);
  pack(gWih_b,  WihBt, 128, 384, 384, 128, 1);
  pack(gWhh_b,  WhhBt, 128, 384, 384, 128, 1);
  pack(beta_W1, W1t,   128, 128, 128, 128, 0);
  pack(beta_W2, W2t,   128, 16,  128, 4,   0);
  pack(cWih,    cWihT, 160, 768, 768, 132, 1);
  pack(cWhh,    cWhhT, 256, 768, 768, 256, 1);
  pack(zeta_W,  zetaT, 384, 256, 384, 256, 0);
  pack(act_W,   actT,  256, 32,  256, 32,  0);
  pack(ups_W,   upsT,  256, 16,  256, 4,   0);
  pack(crit_W,  critT, 256, 16,  256, 1,   0);

  embed_roll_kernel<<<8192 * 32, 128, 0, stream>>>(lines, embedW, X);

  gru_scan_kernel<<<128, 1024, 0, stream>>>(X, WihFt, WhhFt, gbih_f, gbhh_f, G16,
                                            0, +1, 0);
  gru_scan_kernel<<<128, 1024, 0, stream>>>(X, WihBt, WhhBt, gbih_b, gbhh_b, G16,
                                            31, -1, 128);

  beta_kernel<<<4096, 512, 0, stream>>>(G16, W1t, beta_b1, W2t, beta_b2, Bbuf);

  pmask_kernel<<<128, 256, 0, stream>>>(Bbuf, Pbuf);

  step_loop_kernel<<<16, 512, 0, stream>>>(lines, obs, D, embedW,
                                           cWihT, cWhhT, cbih, cbhh,
                                           zetaT, zeta_b, actT, act_b,
                                           upsT, ups_b, critT, crit_b,
                                           Pbuf, out);
}
